// GRU_Decoder_76828374990972
// MI455X (gfx1250) — matrix operand order repaired and checked by reading, not compiled, not run on hardware
//
#include <hip/hip_runtime.h>
#include <hip/hip_bf16.h>

// ---------------------------------------------------------------------------
// GRU decoder, B=32 N=2048 H=256 D_OUT=64 T=12, b = B*N = 65536 rows.
// 64 rows per workgroup, state resident in LDS for all 12 steps, all GEMMs via
// v_wmma_f32_16x16x32_f16, fp32 master state for the recurrence.
// K-loops NOT unrolled (full unroll spilled B fragments to scratch, round 1).
// Activations: v_exp_f32 + v_rcp_f32.  Epilogue r/z split is a single scalar
// branch on readfirstlane(wave) instead of per-lane EXEC masking (round 3).
// ---------------------------------------------------------------------------

typedef __attribute__((ext_vector_type(16))) _Float16 v16h;
typedef __attribute__((ext_vector_type(8)))  float    v8f;

namespace {
constexpr int H    = 256;
constexpr int H2   = 512;
constexpr int DOUT = 64;
constexpr int T    = 12;
constexpr int ROWS = 64;              // rows per workgroup
constexpr int NROW = 65536;           // B*N
constexpr int HCS  = 260;             // s_hc row stride (floats, 1040B, 16B-aligned)
constexpr int XHS  = 520;             // s_xh row stride (halves, 1040B, 16B-aligned)
constexpr int ZS   = 260;             // s_z  row stride (floats)
}

union Frag { uint4 q[2]; v16h h; };

// A fragment (16x32 f16, MxK) from LDS.  ISA layout: lane%16 = M row;
// lanes<16 hold K = k0+0..7 and k0+16..23, lanes>=16 hold K = k0+8..15, k0+24..31.
__device__ __forceinline__ v16h load_a_lds(const _Float16* tile_base, int lane, int k0) {
    const int m  = lane & 15;
    const int kk = k0 + ((lane >> 4) << 3);
    const _Float16* p = tile_base + m * XHS + kk;
    Frag f;
    f.q[0] = *(const uint4*)(p);        // K = kk .. kk+7
    f.q[1] = *(const uint4*)(p + 16);   // K = kk+16 .. kk+23
    return f.h;
}

// B fragment (32x16 f16, KxN) from row-major f16 weight W[n][k] (Kdim = row len).
// ISA layout: lane%16 = N column; lanes<16 hold K = k0+0..7 and k0+16..23, lanes>=16 K = k0+8..15, k0+24..31.
// Column n of B == row n of W -> two runs of 8 contiguous halves per lane (kk.. and kk+16..).
__device__ __forceinline__ v16h load_b_glb(const _Float16* __restrict__ W, int Kdim,
                                           int n0, int lane, int k0) {
    const int n  = n0 + (lane & 15);
    const int kk = k0 + ((lane >> 4) << 3);
    const _Float16* p = W + (long)n * Kdim + kk;
    Frag f;
    f.q[0] = *(const uint4*)(p);        // K = kk .. kk+7
    f.q[1] = *(const uint4*)(p + 16);   // K = kk+16 .. kk+23
    return f.h;
}

__device__ __forceinline__ v8f wmma_f16(v16h a, v16h b, v8f c) {
    return __builtin_amdgcn_wmma_f32_16x16x32_f16(false, a, false, b, (short)0, c,
                                                  false, false);
}

// Hardware v_rcp_f32 (~1 ulp) -- gate activations do not need IEEE division.
__device__ __forceinline__ float fast_rcp(float x) {
    return __builtin_amdgcn_rcpf(x);
}
__device__ __forceinline__ float fast_sigmoid(float x) {
    return fast_rcp(1.0f + __expf(-x));          // v_exp + v_add + v_rcp
}
__device__ __forceinline__ float fast_tanh(float x) {
    const float e = __expf(-2.0f * x);
    return (1.0f - e) * fast_rcp(1.0f + e);      // v_exp + 2 add + v_rcp + mul
}

// Convert fp32 master state -> f16 A-operand panel s_xh[:, 0:H] (packed b32 stores).
__device__ __forceinline__ void convert_hc(const float* s_hc, _Float16* s_xh, int tid) {
    for (int j = tid; j < ROWS * H / 2; j += 256) {
        const int r  = j >> 7;            // 128 half-pairs per row
        const int c2 = (j & 127) << 1;
        const float x0 = s_hc[r * HCS + c2];
        const float x1 = s_hc[r * HCS + c2 + 1];
        union { _Float16 h[2]; unsigned u; } pk;
        pk.h[0] = (_Float16)x0;
        pk.h[1] = (_Float16)x1;
        *(unsigned*)&s_xh[r * XHS + c2] = pk.u;
    }
}

// ---------------------------------------------------------------------------
// Weight prep: fold the duplicated-input gate weight, convert everything to f16.
//   wg[n][k] = gate_w[n][k] + gate_w[n][k+H]   (512 x 256)
//   wo[n][k] = out_w[n][k]                     (256 x 512)
//   wp[n][k] = proj_w[n][k]                    ( 64 x 256)
// ---------------------------------------------------------------------------
__global__ void gru_prep_weights(const float* __restrict__ gate_w,
                                 const float* __restrict__ out_w,
                                 const float* __restrict__ proj_w,
                                 _Float16* __restrict__ wg,
                                 _Float16* __restrict__ wo,
                                 _Float16* __restrict__ wp) {
    const int i = blockIdx.x * 256 + threadIdx.x;
    if (i < H2 * H) {                       // 131072
        const int n = i >> 8, k = i & 255;
        wg[i] = (_Float16)(gate_w[n * H2 + k] + gate_w[n * H2 + H + k]);
    }
    if (i < H * H2) {                       // 131072
        wo[i] = (_Float16)out_w[i];
    }
    if (i < DOUT * H) {                     // 16384
        wp[i] = (_Float16)proj_w[i];
    }
}

// ---------------------------------------------------------------------------
// Main recurrent kernel. 256 threads (8 waves), 64 rows / block, 1024 blocks.
// ---------------------------------------------------------------------------
__global__ __launch_bounds__(256, 1)
void gru_decoder_kernel(const float* __restrict__ h_in,
                        const float* __restrict__ gate_b,
                        const float* __restrict__ out_b,
                        const float* __restrict__ proj_b,
                        const _Float16* __restrict__ wg,
                        const _Float16* __restrict__ wo,
                        const _Float16* __restrict__ wp,
                        float* __restrict__ out) {
    __shared__ float    s_hc[ROWS * HCS];   // fp32 master state          (~65 KB)
    __shared__ _Float16 s_xh[ROWS * XHS];   // f16 [hc | r*hc] A panel    (~65 KB)
    __shared__ float    s_z [ROWS * ZS];    // fp32 update gate z         (~65 KB)

    const int  tid     = threadIdx.x;
    const int  lane    = tid & 31;
    // wave index is uniform across the wave; pin it to an SGPR so all derived
    // tile bases / branches are provably scalar.
    const int  swave   = __builtin_amdgcn_readfirstlane(tid >> 5);
    const long rowBase = (long)blockIdx.x * ROWS;

    // ---- load hidden-state tile: 64 x 256 fp32, float4 vectors ----
    for (int j = tid; j < ROWS * H / 4; j += 256) {
        const int r  = j >> 6;              // 64 float4 per row
        const int c4 = j & 63;
        const float4 v = ((const float4*)(h_in + (rowBase + r) * H))[c4];
        *(float4*)&s_hc[r * HCS + c4 * 4] = v;
    }
    __syncthreads();
    convert_hc(s_hc, s_xh, tid);
    __syncthreads();

    for (int t = 0; t < T; ++t) {
        // ================= GEMM1: gates = hc @ Wg^T  (64x256 . 256x512) =====
        // wave w owns ntiles 4w..4w+3 (of 32) x all 4 mtiles; B reused 4x.
        {
            v8f acc[4][4];
            const v8f vzero = 0.0f;
            #pragma unroll
            for (int n = 0; n < 4; ++n)
                #pragma unroll
                for (int m = 0; m < 4; ++m) acc[n][m] = vzero;

            #pragma unroll 1
            for (int kb = 0; kb < 8; ++kb) {
                const int k0 = kb * 32;
                v16h a[4], b[4];
                #pragma unroll
                for (int m = 0; m < 4; ++m)
                    a[m] = load_a_lds(s_xh + m * 16 * XHS, lane, k0);
                #pragma unroll
                for (int n = 0; n < 4; ++n)
                    b[n] = load_b_glb(wg, H, (swave * 4 + n) * 16, lane, k0);
                #pragma unroll
                for (int n = 0; n < 4; ++n)
                    #pragma unroll
                    for (int m = 0; m < 4; ++m)
                        acc[n][m] = wmma_f16(a[m], b[n], acc[n][m]);
            }
            // Epilogue. waves 0..3 own gate cols 0..255 (r), waves 4..7 own
            // 256..511 (z).  Single scalar branch -- no per-lane EXEC masking.
            if (swave < 4) {
                // r-gates: s_xh[:, H:2H] = (f16)(sigmoid(.) * hc)
                #pragma unroll
                for (int n = 0; n < 4; ++n) {
                    const int gcol = (swave * 4 + n) * 16 + (lane & 15);
                    const float gb = gate_b[gcol];
                    #pragma unroll
                    for (int m = 0; m < 4; ++m) {
                        #pragma unroll
                        for (int v = 0; v < 8; ++v) {
                            const int r = m * 16 + v + ((lane >> 4) << 3);
                            const float g  = fast_sigmoid(acc[n][m][v] + gb);
                            const float hv = s_hc[r * HCS + gcol];
                            s_xh[r * XHS + H + gcol] = (_Float16)(g * hv);
                        }
                    }
                }
            } else {
                // z-gates: s_z = sigmoid(.)
                #pragma unroll
                for (int n = 0; n < 4; ++n) {
                    const int gcol = (swave * 4 + n) * 16 + (lane & 15); // 256..511
                    const int zcol = gcol - H;
                    const float gb = gate_b[gcol];
                    #pragma unroll
                    for (int m = 0; m < 4; ++m) {
                        #pragma unroll
                        for (int v = 0; v < 8; ++v) {
                            const int r = m * 16 + v + ((lane >> 4) << 3);
                            s_z[r * ZS + zcol] = fast_sigmoid(acc[n][m][v] + gb);
                        }
                    }
                }
            }
        }
        __syncthreads();

        // ================= GEMM2: h_new = [hc|r*hc] @ Wo^T (64x512 . 512x256)
        // wave w owns ntiles 2w..2w+1 (of 16) x all 4 mtiles.
        {
            v8f acc[2][4];
            const v8f vzero = 0.0f;
            #pragma unroll
            for (int n = 0; n < 2; ++n)
                #pragma unroll
                for (int m = 0; m < 4; ++m) acc[n][m] = vzero;

            #pragma unroll 1
            for (int kb = 0; kb < 16; ++kb) {
                const int k0 = kb * 32;
                v16h a[4], b[2];
                #pragma unroll
                for (int m = 0; m < 4; ++m)
                    a[m] = load_a_lds(s_xh + m * 16 * XHS, lane, k0);
                #pragma unroll
                for (int n = 0; n < 2; ++n)
                    b[n] = load_b_glb(wo, H2, (swave * 2 + n) * 16, lane, k0);
                #pragma unroll
                for (int n = 0; n < 2; ++n)
                    #pragma unroll
                    for (int m = 0; m < 4; ++m)
                        acc[n][m] = wmma_f16(a[m], b[n], acc[n][m]);
            }
            // epilogue: hn = z*hc + (1-z)*tanh(.)  -> fp32 master state
            #pragma unroll
            for (int n = 0; n < 2; ++n) {
                const int col = (swave * 2 + n) * 16 + (lane & 15);
                const float ob = out_b[col];
                #pragma unroll
                for (int m = 0; m < 4; ++m) {
                    #pragma unroll
                    for (int v = 0; v < 8; ++v) {
                        const int r = m * 16 + v + ((lane >> 4) << 3);
                        const float hnew = fast_tanh(acc[n][m][v] + ob);
                        const float z  = s_z[r * ZS + col];
                        const float hv = s_hc[r * HCS + col];
                        s_hc[r * HCS + col] = z * hv + (1.0f - z) * hnew;
                    }
                }
            }
        }
        __syncthreads();

        // refresh f16 panel (feeds both the projection below and next GEMM1)
        convert_hc(s_hc, s_xh, tid);
        __syncthreads();

        // ================= GEMM3: out = hn @ Wp^T  (64x256 . 256x64) ========
        // 4 ntiles x 4 mtiles = 16 tiles; wave w: ntile = w>>1, mtiles 2(w&1)..+1.
        {
            const int ntile = swave >> 1;
            const int mBase = (swave & 1) * 2;
            v8f acc[2];
            const v8f vzero = 0.0f;
            acc[0] = vzero; acc[1] = vzero;

            #pragma unroll 1
            for (int kb = 0; kb < 8; ++kb) {
                const int k0 = kb * 32;
                const v16h b  = load_b_glb(wp, H, ntile * 16, lane, k0);
                const v16h a0 = load_a_lds(s_xh + (mBase + 0) * 16 * XHS, lane, k0);
                const v16h a1 = load_a_lds(s_xh + (mBase + 1) * 16 * XHS, lane, k0);
                acc[0] = wmma_f16(a0, b, acc[0]);
                acc[1] = wmma_f16(a1, b, acc[1]);
            }
            const int col = ntile * 16 + (lane & 15);
            const float pb = proj_b[col];
            #pragma unroll
            for (int m = 0; m < 2; ++m) {
                #pragma unroll
                for (int v = 0; v < 8; ++v) {
                    const int  r  = (mBase + m) * 16 + v + ((lane >> 4) << 3);
                    const long gr = rowBase + r;
                    out[(gr * T + t) * DOUT + col] = acc[m][v] + pb;
                }
            }
        }
        // No barrier needed: next GEMM1 reads s_xh[:, :H] (stable) and writes
        // s_xh[:, H:] / s_z, disjoint from the projection's reads.
    }
}

// ---------------------------------------------------------------------------
extern "C" void kernel_launch(void* const* d_in, const int* in_sizes, int n_in,
                              void* d_out, int out_size, void* d_ws, size_t ws_size,
                              hipStream_t stream) {
    (void)in_sizes; (void)n_in; (void)out_size; (void)ws_size;
    // inputs: [0]=x(T, unused; compile-time 12), [1]=h, [2]=gate_w, [3]=gate_b,
    //         [4]=out_w, [5]=out_b, [6]=proj_w, [7]=proj_b
    const float* h      = (const float*)d_in[1];
    const float* gate_w = (const float*)d_in[2];
    const float* gate_b = (const float*)d_in[3];
    const float* out_w  = (const float*)d_in[4];
    const float* out_b  = (const float*)d_in[5];
    const float* proj_w = (const float*)d_in[6];
    const float* proj_b = (const float*)d_in[7];

    _Float16* wg = (_Float16*)d_ws;          // 512*256
    _Float16* wo = wg + H2 * H;              // 256*512
    _Float16* wp = wo + H * H2;              //  64*256

    gru_prep_weights<<<512, 256, 0, stream>>>(gate_w, out_w, proj_w, wg, wo, wp);

    gru_decoder_kernel<<<NROW / ROWS, 256, 0, stream>>>(
        h, gate_b, out_b, proj_b, wg, wo, wp, (float*)d_out);
}
